// ObliviousForest_87531433492895
// MI455X (gfx1250) — compile-verified, matrix operand order checked by reading, not run on hardware
//
#include <hip/hip_runtime.h>
#include <hip/hip_bf16.h>

// ---------------------------------------------------------------------------
// ObliviousForest on MI455X (gfx1250, wave32, WMMA)
//
// Stage 1: facts      -> f_aug (B x FIN) f16
// Stage 2: sparsemax  -> sel   (NROWS x FIN) f16
//                        (row cached in LDS; bisection for tau, no sort)
// Stage 3: WMMA GEMM  -> P = f_aug @ sel^T  (B x NROWS) f32
//                        32x64 wave tile = 2x4 fragments of v_wmma_f32_16x16x32_f16
// Stage 4: forest     -> leaf-prob doubling + leaf_value reduction -> out (B x 16)
//                        (16 batch rows per block to reuse leaf_value from L2)
// ---------------------------------------------------------------------------

#define INPUT_DIM 256
#define N_TREES   512
#define DEPTH     6
#define N_THRESH  6
#define F0        (INPUT_DIM * N_THRESH)   // 1536
#define FIN       (2 * F0)                 // 3072
#define LEAVES    64
#define OUT_DIM   16
#define BATCH     4096
#define NROWS     (N_TREES * DEPTH)        // 3072

typedef __attribute__((ext_vector_type(16))) _Float16 v16h;
typedef __attribute__((ext_vector_type(8)))  _Float16 v8h;
typedef __attribute__((ext_vector_type(8)))  float    v8f;

// ---------------------------------------------------------------------------
// Stage 1: f = sigmoid((x - thresh) * slope); f_aug = [f, 1-f] as f16
// ---------------------------------------------------------------------------
__global__ void of_facts_kernel(const float* __restrict__ x,
                                const float* __restrict__ th,
                                const float* __restrict__ log_slope,
                                _Float16* __restrict__ faug) {
    int idx = blockIdx.x * blockDim.x + threadIdx.x;   // over B * F0
    if (idx >= BATCH * F0) return;
    int b = idx / F0;
    int j = idx - b * F0;
    int d = j / N_THRESH;
    int k = j - d * N_THRESH;
    float slope = __expf(log_slope[0]);
    float z = (x[b * INPUT_DIM + d] - th[d * N_THRESH + k]) * slope;
    float f = 1.0f / (1.0f + __expf(-z));
    faug[(size_t)b * FIN + j]      = (_Float16)f;
    faug[(size_t)b * FIN + F0 + j] = (_Float16)(1.0f - f);
}

// ---------------------------------------------------------------------------
// Stage 2: sparsemax per row via bisection on tau.
//   sum_i max(z_i - tau, 0) is monotone decreasing in tau;
//   tau in [rowmax - 1, rowmax]. 40 bisection steps -> f32 precision.
// One 256-thread block per row. The 12 KB row is staged into LDS once, so the
// 41 reduction passes run out of LDS instead of re-reading L2.
// ---------------------------------------------------------------------------
__global__ void of_sparsemax_kernel(const float* __restrict__ logits,
                                    _Float16* __restrict__ sel) {
    __shared__ float zs[FIN];    // 12 KB row cache
    __shared__ float red[256];
    const int row = blockIdx.x;
    const int t   = threadIdx.x;
    const float* __restrict__ z = logits + (size_t)row * FIN;

    // Stage the row into LDS (single global pass)
    for (int i = t; i < FIN; i += 256) zs[i] = z[i];
    __syncthreads();

    // Row max
    float m = -3.0e38f;
    for (int i = t; i < FIN; i += 256) m = fmaxf(m, zs[i]);
    red[t] = m; __syncthreads();
    for (int s = 128; s > 0; s >>= 1) {
        if (t < s) red[t] = fmaxf(red[t], red[t + s]);
        __syncthreads();
    }
    const float rowmax = red[0];
    __syncthreads();

    // Bisection for tau
    float lo = rowmax - 1.0f;   // sum >= 1 here
    float hi = rowmax;          // sum == 0 here
    for (int it = 0; it < 40; ++it) {
        const float mid = 0.5f * (lo + hi);
        float s = 0.0f;
        for (int i = t; i < FIN; i += 256) s += fmaxf(zs[i] - mid, 0.0f);
        red[t] = s; __syncthreads();
        for (int s2 = 128; s2 > 0; s2 >>= 1) {
            if (t < s2) red[t] += red[t + s2];
            __syncthreads();
        }
        const float total = red[0];   // uniform across block
        __syncthreads();
        if (total > 1.0f) lo = mid; else hi = mid;
    }
    const float tau = 0.5f * (lo + hi);

    _Float16* __restrict__ srow = sel + (size_t)row * FIN;
    for (int i = t; i < FIN; i += 256)
        srow[i] = (_Float16)fmaxf(zs[i] - tau, 0.0f);
}

// ---------------------------------------------------------------------------
// Stage 3: P[b, n] = sum_k faug[b, k] * sel[n, k]   (M=4096, N=3072, K=3072)
//
// Each wave computes a 32x64 macro-tile (2x4 fragments), K-loop step 32:
//   12 b128 loads (6 fragments) feed 8 v_wmma_f32_16x16x32_f16 per step.
//   ~21.3 flops per L2 byte; total L2 traffic ~3.6 GB for the 77 GFLOP GEMM.
//
// A-operand lane layout (ISA 7.12.2, 16-bit A 16x32):
//   lanes 0-15:  row M = lane,    K = {0..7, 16..23}
//   lanes 16-31: row M = lane-16, K = {8..15, 24..31}
// B operand holds B^T (= sel rows) in the same layout.
// C/D layout: acc[v] -> (M = v + 8*half, N = lane&15).
// ---------------------------------------------------------------------------
#define WT_M         32                    // wave tile M edge
#define WT_N         64                    // wave tile N edge
#define WTILES_M     (BATCH / WT_M)        // 128
#define WTILES_N     (NROWS / WT_N)        // 48
#define WAVES_PER_BLOCK 8

__global__ void
__attribute__((amdgpu_flat_work_group_size(256, 256), amdgpu_waves_per_eu(1)))
of_gemm_kernel(const _Float16* __restrict__ A,
               const _Float16* __restrict__ Bm,
               float* __restrict__ P) {
    const int wave = threadIdx.x >> 5;
    const int lane = threadIdx.x & 31;
    const int wt   = blockIdx.x * WAVES_PER_BLOCK + wave;
    if (wt >= WTILES_M * WTILES_N) return;
    const int mt = wt / WTILES_N;
    const int nt = wt - mt * WTILES_N;
    const int m0 = mt * WT_M;
    const int n0 = nt * WT_N;
    const int half = lane >> 4;       // which K-half this lane supplies
    const int r    = lane & 15;       // row within fragment

    const _Float16* __restrict__ arow[2];
    const _Float16* __restrict__ brow[4];
#pragma unroll
    for (int i = 0; i < 2; ++i)
        arow[i] = A + (size_t)(m0 + i * 16 + r) * FIN;
#pragma unroll
    for (int j = 0; j < 4; ++j)
        brow[j] = Bm + (size_t)(n0 + j * 16 + r) * FIN;

    v8f acc[2][4] = {};
    for (int k0 = 0; k0 < FIN; k0 += 32) {
        const int kb = k0 + half * 8;

        v16h af[2], bf[4];
#pragma unroll
        for (int i = 0; i < 2; ++i) {
            v8h lo = *(const v8h*)(arow[i] + kb);
            v8h hi = *(const v8h*)(arow[i] + kb + 16);
#pragma unroll
            for (int e = 0; e < 8; ++e) {
                af[i][e]     = lo[e];
                af[i][8 + e] = hi[e];
            }
        }
#pragma unroll
        for (int j = 0; j < 4; ++j) {
            v8h lo = *(const v8h*)(brow[j] + kb);
            v8h hi = *(const v8h*)(brow[j] + kb + 16);
#pragma unroll
            for (int e = 0; e < 8; ++e) {
                bf[j][e]     = lo[e];
                bf[j][8 + e] = hi[e];
            }
        }
#pragma unroll
        for (int i = 0; i < 2; ++i) {
#pragma unroll
            for (int j = 0; j < 4; ++j) {
                acc[i][j] = __builtin_amdgcn_wmma_f32_16x16x32_f16(
                    /*neg_a=*/false, af[i], /*neg_b=*/false, bf[j],
                    /*c_mod=*/(short)0, acc[i][j],
                    /*reuse_a=*/false, /*reuse_b=*/false);
            }
        }
    }

#pragma unroll
    for (int i = 0; i < 2; ++i) {
#pragma unroll
        for (int j = 0; j < 4; ++j) {
            float* prow = P + (size_t)(m0 + i * 16 + half * 8) * NROWS
                            + (n0 + j * 16 + r);
#pragma unroll
            for (int v = 0; v < 8; ++v)
                prow[(size_t)v * NROWS] = acc[i][j][v];
        }
    }
}

// ---------------------------------------------------------------------------
// Stage 4: clip p; leaf probs via bit-indexed products; reduce over trees.
// Each block processes 16 batch rows so leaf_value (2 MB) is re-used across
// rows: L2 leaf traffic drops from ~8 GB (1 row/block) to ~512 MB.
// Per row: each thread handles 2 trees, then a 16 KB LDS tree-reduce.
// ---------------------------------------------------------------------------
#define ROWS_PER_BLOCK 16

__global__ void of_forest_kernel(const float* __restrict__ P,
                                 const float* __restrict__ leaf,
                                 float* __restrict__ out) {
    __shared__ float red[256 * OUT_DIM];   // 16 KB
    const int t  = threadIdx.x;
    const int b0 = blockIdx.x * ROWS_PER_BLOCK;

    for (int bb = 0; bb < ROWS_PER_BLOCK; ++bb) {
        const int b = b0 + bb;

        float acc[OUT_DIM];
#pragma unroll
        for (int c = 0; c < OUT_DIM; ++c) acc[c] = 0.0f;

        for (int tr = t; tr < N_TREES; tr += 256) {
            float pd[DEPTH], qd[DEPTH];
            const float* __restrict__ prow = P + (size_t)b * NROWS + tr * DEPTH;
#pragma unroll
            for (int d = 0; d < DEPTH; ++d) {
                float p = prow[d];
                p = fminf(fmaxf(p, 1e-6f), 1.0f - 1e-6f);
                pd[d] = p;
                qd[d] = 1.0f - p;
            }
            const float* __restrict__ lv = leaf + (size_t)tr * LEAVES * OUT_DIM;
#pragma unroll
            for (int l = 0; l < LEAVES; ++l) {
                float w = ((l >> 0) & 1) ? pd[0] : qd[0];
#pragma unroll
                for (int d = 1; d < DEPTH; ++d)
                    w *= ((l >> d) & 1) ? pd[d] : qd[d];
#pragma unroll
                for (int c = 0; c < OUT_DIM; ++c)
                    acc[c] = fmaf(w, lv[l * OUT_DIM + c], acc[c]);
            }
        }

#pragma unroll
        for (int c = 0; c < OUT_DIM; ++c) red[t * OUT_DIM + c] = acc[c];
        __syncthreads();
        for (int s = 128; s > 0; s >>= 1) {
            if (t < s) {
#pragma unroll
                for (int c = 0; c < OUT_DIM; ++c)
                    red[t * OUT_DIM + c] += red[(t + s) * OUT_DIM + c];
            }
            __syncthreads();
        }
        if (t < OUT_DIM) out[(size_t)b * OUT_DIM + t] = red[t];
        __syncthreads();   // protect red[] before next row overwrites it
    }
}

// ---------------------------------------------------------------------------
// Launcher. Workspace layout (all 256B-aligned by construction):
//   [0)                       sel  f16  NROWS*FIN        = 18,874,368 B
//   [selBytes)                faug f16  BATCH*FIN        = 25,165,824 B
//   [selBytes+faugBytes)      P    f32  BATCH*NROWS      = 50,331,648 B
// Total ~ 90 MB (fits the 192 MB L2 alongside inputs).
// ---------------------------------------------------------------------------
extern "C" void kernel_launch(void* const* d_in, const int* in_sizes, int n_in,
                              void* d_out, int out_size, void* d_ws, size_t ws_size,
                              hipStream_t stream) {
    (void)in_sizes; (void)n_in; (void)out_size; (void)ws_size;

    const float* x          = (const float*)d_in[0];
    const float* thresholds = (const float*)d_in[1];
    const float* log_slope  = (const float*)d_in[2];
    const float* sel_logits = (const float*)d_in[3];
    const float* leaf_value = (const float*)d_in[4];
    float* out              = (float*)d_out;

    char* ws = (char*)d_ws;
    const size_t selBytes  = (size_t)NROWS * FIN * sizeof(_Float16);
    const size_t faugBytes = (size_t)BATCH * FIN * sizeof(_Float16);
    _Float16* sel  = (_Float16*)ws;
    _Float16* faug = (_Float16*)(ws + selBytes);
    float*    P    = (float*)(ws + selBytes + faugBytes);

    // Stage 1: fact bank
    {
        int total = BATCH * F0;
        dim3 grid((total + 255) / 256), block(256);
        of_facts_kernel<<<grid, block, 0, stream>>>(x, thresholds, log_slope, faug);
    }
    // Stage 2: sparsemax rows (LDS-cached)
    {
        dim3 grid(NROWS), block(256);
        of_sparsemax_kernel<<<grid, block, 0, stream>>>(sel_logits, sel);
    }
    // Stage 3: WMMA GEMM  P = faug @ sel^T  (32x64 per wave)
    {
        int wtiles = WTILES_M * WTILES_N;              // 6144 wave tiles
        dim3 grid(wtiles / WAVES_PER_BLOCK), block(32 * WAVES_PER_BLOCK);
        of_gemm_kernel<<<grid, block, 0, stream>>>(faug, sel, P);
    }
    // Stage 4: forest reduction (16 rows per block)
    {
        dim3 grid(BATCH / ROWS_PER_BLOCK), block(256);
        of_forest_kernel<<<grid, block, 0, stream>>>(P, leaf_value, out);
    }
}
